// GINEncoder_18047452577893
// MI455X (gfx1250) — matrix operand order repaired and checked by reading, not compiled, not run on hardware
//
#include <hip/hip_runtime.h>
#include <hip/hip_bf16.h>

#define NN 50000
#define EE 640000
#define HH 128
#define MLPH 256
#define GG 64
#define LL 4

typedef __attribute__((ext_vector_type(16))) _Float16 v16h;
typedef __attribute__((ext_vector_type(8)))  _Float16 v8h;
typedef __attribute__((ext_vector_type(4)))  _Float16 v4h;
typedef __attribute__((ext_vector_type(8)))  float    v8f;
typedef __attribute__((ext_vector_type(4))) unsigned int u32x4;
typedef __attribute__((ext_vector_type(8))) int          i32x8;
typedef __attribute__((ext_vector_type(4))) int          i32x4;

// ---------------- CDNA5 async-copy (global -> LDS, ASYNCcnt) ----------------
#if __has_builtin(__builtin_amdgcn_global_load_async_to_lds_b128)
#define HAVE_ASYNC 1
static __device__ inline void async_cp16(const _Float16* g, _Float16* l) {
    __builtin_amdgcn_global_load_async_to_lds_b128(
        (__attribute__((address_space(1))) i32x4*)(_Float16*)g,
        (__attribute__((address_space(3))) i32x4*)l, 0, 0);
}
#if __has_builtin(__builtin_amdgcn_s_wait_asynccnt)
#define ASYNC_WAIT0() __builtin_amdgcn_s_wait_asynccnt(0)
#else
#define ASYNC_WAIT0() asm volatile("s_wait_asynccnt 0x0" ::: "memory")
#endif
#else
#define HAVE_ASYNC 0
#define ASYNC_WAIT0() ((void)0)
#endif

// ---------------- CDNA5 Tensor Data Mover (TENSORcnt) ----------------
#if __has_builtin(__builtin_amdgcn_tensor_load_to_lds)
#define HAVE_TDM 1

static __device__ inline unsigned lds_byte_offset(const void* p) {
    return (unsigned)(size_t)(const __attribute__((address_space(3))) void*)p;
}

// 1-D contiguous tile of f16: nElems halves from gsrc into LDS at ldsOff.
static __device__ inline void tdm_load_1d_f16(const _Float16* gsrc, unsigned ldsOff,
                                              unsigned nElems) {
    unsigned long long ga = (unsigned long long)(size_t)gsrc;
    u32x4 g0;
    g0[0] = 1u;                                     // count=1, user descriptor
    g0[1] = ldsOff;                                 // lds_addr (bytes)
    g0[2] = (unsigned)(ga & 0xFFFFFFFFu);           // global_addr[31:0]
    g0[3] = (unsigned)((ga >> 32) & 0x01FFFFFFu)    // global_addr[56:32]
            | (2u << 30);                           // type=2 ("image")
    i32x8 g1;
    g1[0] = (int)(1u << 16);                        // data_size=1 (2 bytes)
    g1[1] = (int)((nElems & 0xFFFFu) << 16);        // tensor_dim0[15:0]
    g1[2] = (int)((nElems >> 16) & 0xFFFFu)         // tensor_dim0[31:16]
            | (int)(1u << 16);                      // tensor_dim1 = 1
    g1[3] = (int)((nElems & 0xFFFFu) << 16);        // tile_dim0 = nElems
    g1[4] = 0;                                      // tile_dim1/2 unused
    g1[5] = (int)nElems;                            // tensor_dim0_stride
    g1[6] = 0;
    g1[7] = 0;
    i32x4 z4 = {0, 0, 0, 0};
    i32x8 z8 = {0, 0, 0, 0, 0, 0, 0, 0};
    __builtin_amdgcn_tensor_load_to_lds(g0, g1, z4, z4, z8, 0);
}
#define TDM_WAIT0() __builtin_amdgcn_s_wait_tensorcnt(0)
#else
#define HAVE_TDM 0
#endif

// ---------------- WMMA fragment helpers (wave32, 16x16x32 f16) ----------------
// A (16x32, 16-bit) per ISA 7.12.2:
//   lanes 0-15 : row M=l16, VGPR0-3 = K 0..7,  VGPR4-7 = K 16..23
//   lanes 16-31: row M=l16, VGPR0-3 = K 8..15, VGPR4-7 = K 24..31
// B (32x16, 16-bit): col N=l16, K order same as A (0..7/16..23 lo, 8..15/24..31 hi).
// C/D (16x16 f32): VGPR r -> row = r + 8*hi, col = l16.

static __device__ inline v16h pack16(v8h lo, v8h hh) {
    v16h r;
#pragma unroll
    for (int i = 0; i < 8; ++i) { r[i] = lo[i]; r[8 + i] = hh[i]; }
    return r;
}

static __device__ inline v8h cvt8(const float* __restrict__ p) {
    float4 a = ((const float4*)p)[0];
    float4 b = ((const float4*)p)[1];
    v8h r;
    r[0] = (_Float16)a.x; r[1] = (_Float16)a.y; r[2] = (_Float16)a.z; r[3] = (_Float16)a.w;
    r[4] = (_Float16)b.x; r[5] = (_Float16)b.y; r[6] = (_Float16)b.z; r[7] = (_Float16)b.w;
    return r;
}

static __device__ inline v16h glb_afrag(const float* __restrict__ A, int row, int lda,
                                        int k0, int hi) {
    const float* p = A + (size_t)row * lda + k0 + hi * 8;
    return pack16(cvt8(p), cvt8(p + 16));
}

static __device__ inline v16h lds_afrag(const _Float16* __restrict__ base, int row0,
                                        int k0, int ld, int l16, int hi) {
    const _Float16* p = base + (size_t)(row0 + l16) * ld + k0 + hi * 8;
    v8h lo = *(const v8h*)p;
    v8h hh = *(const v8h*)(p + 16);
    return pack16(lo, hh);
}

static __device__ inline v16h lds_bfrag(const _Float16* __restrict__ Wc, int lane, int n0) {
    const _Float16* p = Wc + ((lane >> 4) * 8) * 128 + n0 + (lane & 15);
    v16h r;
#pragma unroll
    for (int i = 0; i < 8; ++i) { r[i] = p[i * 128]; r[8 + i] = p[(16 + i) * 128]; }
    return r;
}

static __device__ inline v8f do_wmma(v16h a, v16h b, v8f c) {
    return __builtin_amdgcn_wmma_f32_16x16x32_f16(false, a, false, b, (short)0, c, false, false);
}

// Issue staging of f16 W chunk [32 x 128] into LDS buffer (async if available).
static __device__ inline void stage_issue(const _Float16* __restrict__ Wh, int ldW, int k0,
                                          int colBase, _Float16* __restrict__ Wc, int tid) {
    int r = tid >> 3;
    int seg = (tid & 7) << 4;
    const _Float16* g = Wh + (size_t)(k0 + r) * ldW + colBase + seg;
    _Float16* l = Wc + r * 128 + seg;
#if HAVE_ASYNC
    async_cp16(g, l);
    async_cp16(g + 8, l + 8);
#else
    *(v8h*)l = *(const v8h*)g;
    *(v8h*)(l + 8) = *(const v8h*)(g + 8);
#endif
}

// ---------------- node-feature GEMM:  C = epi(A @ Wh + bias) ----------------
// EPI 0: +bias | EPI 1: +bias,BN,ReLU | EPI 2: +bias,BN,ReLU,+=res
// Block 256 thr = 8 waves; tile 128 rows x 128 cols; grid.y covers KOUT/128.
// Double-buffered LDS W chunks; async DMA overlaps WMMA compute.
template <int EPI>
__global__ __launch_bounds__(256) void k_gemm(
    const float* __restrict__ A, const _Float16* __restrict__ Wh,
    const float* __restrict__ bias,
    const float* __restrict__ bng, const float* __restrict__ bnb,
    const float* __restrict__ bnm, const float* __restrict__ bnv,
    float* __restrict__ C, const float* __restrict__ res,
    int nRows, int KIN, int KOUT) {
    __shared__ _Float16 Wc[2][32 * 128];

    const int tid  = threadIdx.x;
    const int lane = tid & 31;
    const int wv   = tid >> 5;
    const int l16  = lane & 15;
    const int hi   = lane >> 4;

    const int rowBase = blockIdx.x * 128 + wv * 16;
    const int colBase = blockIdx.y * 128;
    int rowA = rowBase + l16;
    if (rowA >= nRows) rowA = nRows - 1;   // clamp; OOB rows are never stored

    v8f acc[8];
#pragma unroll
    for (int i = 0; i < 8; ++i) acc[i] = (v8f)0.0f;

    const int nK = KIN >> 5;
    stage_issue(Wh, KOUT, 0, colBase, Wc[0], tid);     // prologue: buffer 0

    for (int kc = 0; kc < nK; ++kc) {
        const int p = kc & 1;
        ASYNC_WAIT0();          // own chunk p landed (no-op on sync path)
        __syncthreads();        // everyone's chunk p landed; all done reading 1-p
        if (kc + 1 < nK)
            stage_issue(Wh, KOUT, (kc + 1) << 5, colBase, Wc[1 - p], tid);
        v16h a = glb_afrag(A, rowA, KIN, kc << 5, hi);
#pragma unroll
        for (int nt = 0; nt < 8; ++nt) {
            v16h b = lds_bfrag(Wc[p], lane, nt * 16);
            acc[nt] = do_wmma(a, b, acc[nt]);
        }
    }

#pragma unroll
    for (int nt = 0; nt < 8; ++nt) {
        const int col = colBase + nt * 16 + l16;
        const float bb = bias[col];
        float sc = 1.0f, sh = 0.0f;
        if (EPI >= 1) {
            float g  = bng[col];
            float be = bnb[col];
            float mu = bnm[col];
            float vv = bnv[col];
            sc = g * rsqrtf(vv + 1e-5f);
            sh = be - mu * sc;
        }
#pragma unroll
        for (int r = 0; r < 8; ++r) {
            const int row = rowBase + r + 8 * hi;
            if (row < nRows) {
                float y = acc[nt][r] + bb;
                if (EPI >= 1) {
                    y = y * sc + sh;
                    y = y > 0.0f ? y : 0.0f;
                }
                if (EPI == 2) y += res[(size_t)row * KOUT + col];
                C[(size_t)row * KOUT + col] = y;
            }
        }
    }
}

// ---------------- weight convert f32 -> f16 (float4 granularity) ----------------
__global__ __launch_bounds__(256) void k_cvt(const float* __restrict__ s,
                                             _Float16* __restrict__ d, int n4) {
    int i = blockIdx.x * blockDim.x + threadIdx.x;
    if (i < n4) {
        float4 v = ((const float4*)s)[i];
        v4h o;
        o[0] = (_Float16)v.x; o[1] = (_Float16)v.y;
        o[2] = (_Float16)v.z; o[3] = (_Float16)v.w;
        ((v4h*)d)[i] = o;
    }
}

// ---------------- copy m := h (float4) ----------------
__global__ __launch_bounds__(256) void k_copy4(const float* __restrict__ src,
                                               float* __restrict__ dst, int n4) {
    int i = blockIdx.x * blockDim.x + threadIdx.x;
    if (i < n4) ((float4*)dst)[i] = ((const float4*)src)[i];
}

// ---------------- edge scatter-add: m[dst] += h[src] ----------------
__global__ __launch_bounds__(256) void k_edges(const float* __restrict__ h,
                                               const int* __restrict__ ei,
                                               float* __restrict__ m, int nE) {
    long tid = (long)blockIdx.x * blockDim.x + threadIdx.x;
    long total = (long)nE * 32;
    if (tid >= total) return;
    int e = (int)(tid >> 5);
    int c = (int)(tid & 31) << 2;
    int s = ei[e];
    int d = ei[nE + e];
    float4 v = *(const float4*)(h + (size_t)s * HH + c);
    float* dp = m + (size_t)d * HH + c;
    atomicAdd(dp + 0, v.x);
    atomicAdd(dp + 1, v.y);
    atomicAdd(dp + 2, v.z);
    atomicAdd(dp + 3, v.w);
}

// ---------------- zero + pool ----------------
__global__ __launch_bounds__(256) void k_zero(float* __restrict__ p, int n) {
    int i = blockIdx.x * blockDim.x + threadIdx.x;
    if (i < n) p[i] = 0.0f;
}

__global__ __launch_bounds__(256) void k_pool(const float* __restrict__ h,
                                              const int* __restrict__ batch,
                                              float* __restrict__ pooled, int n) {
    long tid = (long)blockIdx.x * blockDim.x + threadIdx.x;
    long total = (long)n * 32;
    if (tid >= total) return;
    int node = (int)(tid >> 5);
    int c = (int)(tid & 31) << 2;
    int g = batch[node];
    float4 v = *(const float4*)(h + (size_t)node * HH + c);
    float* dp = pooled + (size_t)g * HH + c;
    atomicAdd(dp + 0, v.x);
    atomicAdd(dp + 1, v.y);
    atomicAdd(dp + 2, v.z);
    atomicAdd(dp + 3, v.w);
}

// ---------------- head: out = relu(pooled@W1+b1)@W2 + b2  (64x128) ----------------
// Weight chunks staged f16 via Tensor Data Mover when available.
__global__ __launch_bounds__(256) void k_head(const float* __restrict__ pooled,
                                              const _Float16* __restrict__ W1h,
                                              const float* __restrict__ b1,
                                              const _Float16* __restrict__ W2h,
                                              const float* __restrict__ b2,
                                              float* __restrict__ out) {
    __shared__ _Float16 Ap[64 * 128];  // pooled, f16
    __shared__ _Float16 Wc[32 * 128];  // weight chunk (4096 halves)
    __shared__ _Float16 T[64 * 128];   // hidden, f16

    const int tid  = threadIdx.x;
    const int lane = tid & 31;
    const int wv   = tid >> 5;
    const int l16  = lane & 15;
    const int hi   = lane >> 4;

    for (int i = tid; i < 64 * 128; i += 256) Ap[i] = (_Float16)pooled[i];
    __syncthreads();

#pragma unroll
    for (int phase = 0; phase < 2; ++phase) {
        const _Float16* Wsrc = phase ? W2h : W1h;
        const float*    bb_v = phase ? b2 : b1;
        const _Float16* Asrc = phase ? T : Ap;

        v8f acc[4];
#pragma unroll
        for (int i = 0; i < 4; ++i) acc[i] = (v8f)0.0f;

        for (int kc = 0; kc < 4; ++kc) {
            const int k0 = kc << 5;
            __syncthreads();
#if HAVE_TDM
            if (wv == 0) {
                tdm_load_1d_f16(Wsrc + (size_t)k0 * 128, lds_byte_offset(Wc), 32 * 128);
                TDM_WAIT0();
            }
#else
            {
                int r = tid >> 3, seg = (tid & 7) << 4;
                const _Float16* g = Wsrc + (size_t)(k0 + r) * 128 + seg;
                _Float16* l = Wc + r * 128 + seg;
                *(v8h*)l = *(const v8h*)g;
                *(v8h*)(l + 8) = *(const v8h*)(g + 8);
            }
#endif
            __syncthreads();
#pragma unroll
            for (int tt = 0; tt < 4; ++tt) {
                int tile = wv + 8 * tt;
                int rt = tile & 3, nt = tile >> 2;
                v16h a = lds_afrag(Asrc, rt * 16, k0, 128, l16, hi);
                v16h b = lds_bfrag(Wc, lane, nt * 16);
                acc[tt] = do_wmma(a, b, acc[tt]);
            }
        }
        __syncthreads();
#pragma unroll
        for (int tt = 0; tt < 4; ++tt) {
            int tile = wv + 8 * tt;
            int rt = tile & 3, nt = tile >> 2;
            int col = nt * 16 + l16;
            float bb = bb_v[col];
#pragma unroll
            for (int r = 0; r < 8; ++r) {
                int row = rt * 16 + r + 8 * hi;
                float y = acc[tt][r] + bb;
                if (phase == 0) {
                    y = y > 0.0f ? y : 0.0f;
                    T[row * 128 + col] = (_Float16)y;
                } else {
                    out[row * 128 + col] = y;
                }
            }
        }
        if (phase == 0) __syncthreads();
    }
}

// ---------------- launcher ----------------
extern "C" void kernel_launch(void* const* d_in, const int* in_sizes, int n_in,
                              void* d_out, int out_size, void* d_ws, size_t ws_size,
                              hipStream_t stream) {
    const float* x     = (const float*)d_in[0];
    const int*   ei    = (const int*)d_in[1];
    const int*   batch = (const int*)d_in[2];
    const float* W0    = (const float*)d_in[3];
    const float* b0    = (const float*)d_in[4];
    const float* mlpW1 = (const float*)d_in[5];
    const float* mlpb1 = (const float*)d_in[6];
    const float* bn1g  = (const float*)d_in[7];
    const float* bn1b  = (const float*)d_in[8];
    const float* bn1m  = (const float*)d_in[9];
    const float* bn1v  = (const float*)d_in[10];
    const float* mlpW2 = (const float*)d_in[11];
    const float* mlpb2 = (const float*)d_in[12];
    const float* ng    = (const float*)d_in[13];
    const float* nb    = (const float*)d_in[14];
    const float* nm    = (const float*)d_in[15];
    const float* nv    = (const float*)d_in[16];
    const float* W1    = (const float*)d_in[17];
    const float* b1    = (const float*)d_in[18];
    const float* W2    = (const float*)d_in[19];
    const float* b2    = (const float*)d_in[20];
    float* out = (float*)d_out;

    float* h      = (float*)d_ws;
    float* m      = h + (size_t)NN * HH;
    float* m2     = m + (size_t)NN * HH;
    float* pooled = m2 + (size_t)NN * MLPH;

    _Float16* W0h  = (_Float16*)(pooled + GG * HH);
    _Float16* W1hA = W0h + (size_t)HH * HH;                 // L*H*MLPH
    _Float16* W2hA = W1hA + (size_t)LL * HH * MLPH;         // L*MLPH*H
    _Float16* Wh1  = W2hA + (size_t)LL * MLPH * HH;
    _Float16* Wh2  = Wh1 + (size_t)HH * HH;

    dim3 blk(256);
    dim3 gRows((NN + 127) / 128, 1);
    dim3 gRows2((NN + 127) / 128, 2);

    // Weights -> f16 once (tiny; keeps GEMM B-staging a raw byte copy)
    k_cvt<<<(HH * HH / 4 + 255) / 256, blk, 0, stream>>>(W0, W0h, HH * HH / 4);
    k_cvt<<<(LL * HH * MLPH / 4 + 255) / 256, blk, 0, stream>>>(mlpW1, W1hA, LL * HH * MLPH / 4);
    k_cvt<<<(LL * MLPH * HH / 4 + 255) / 256, blk, 0, stream>>>(mlpW2, W2hA, LL * MLPH * HH / 4);
    k_cvt<<<(HH * HH / 4 + 255) / 256, blk, 0, stream>>>(W1, Wh1, HH * HH / 4);
    k_cvt<<<(HH * HH / 4 + 255) / 256, blk, 0, stream>>>(W2, Wh2, HH * HH / 4);

    // h = x @ W0 + b0
    k_gemm<0><<<gRows, blk, 0, stream>>>(x, W0h, b0, nullptr, nullptr, nullptr, nullptr,
                                         h, nullptr, NN, 128, 128);

    const int copy4 = NN * HH / 4;
    const long eThreads = (long)EE * 32;
    const long pThreads = (long)NN * 32;

    for (int l = 0; l < LL; ++l) {
        // m = h + scatter_add(h[src] -> dst)
        k_copy4<<<(copy4 + 255) / 256, blk, 0, stream>>>(h, m, copy4);
        k_edges<<<(unsigned)((eThreads + 255) / 256), blk, 0, stream>>>(h, ei, m, EE);
        // m2 = relu(bn(m @ mlpW1 + b))
        k_gemm<1><<<gRows2, blk, 0, stream>>>(m, W1hA + (size_t)l * HH * MLPH,
                                              mlpb1 + (size_t)l * MLPH,
                                              bn1g + (size_t)l * MLPH, bn1b + (size_t)l * MLPH,
                                              bn1m + (size_t)l * MLPH, bn1v + (size_t)l * MLPH,
                                              m2, nullptr, NN, 128, 256);
        // h = h + relu(bn(m2 @ mlpW2 + b))
        k_gemm<2><<<gRows, blk, 0, stream>>>(m2, W2hA + (size_t)l * MLPH * HH,
                                             mlpb2 + (size_t)l * HH,
                                             ng + (size_t)l * HH, nb + (size_t)l * HH,
                                             nm + (size_t)l * HH, nv + (size_t)l * HH,
                                             h, h, NN, 256, 128);
    }

    // pooled = segment_sum(h, batch)
    k_zero<<<(GG * HH + 255) / 256, blk, 0, stream>>>(pooled, GG * HH);
    k_pool<<<(unsigned)((pThreads + 255) / 256), blk, 0, stream>>>(h, batch, pooled, NN);

    // out = relu(pooled @ W1 + b1) @ W2 + b2
    k_head<<<1, blk, 0, stream>>>(pooled, Wh1, b1, Wh2, b2, out);
}
